// GAT_87531433493194
// MI455X (gfx1250) — matrix operand order repaired and checked by reading, not compiled, not run on hardware
//
#include <hip/hip_runtime.h>
#include <hip/hip_bf16.h>
#include <cstdint>

typedef __attribute__((ext_vector_type(16))) _Float16 v16h;
typedef __attribute__((ext_vector_type(8)))  float    v8f;

#define WMMA_F16(A, B, C) \
    __builtin_amdgcn_wmma_f32_16x16x32_f16(false, (A), false, (B), (short)0, (C), false, false)

#define N_NODES 8192
#define D_IN    128
#define D_HID   64
#define WORDS_PER_ROW (N_NODES / 32)   // 256 packed adjacency words per row
#define NEG_BIG (-1e30f)

// ---------------------------------------------------------------------------
// Kernel A: q/k/v projections. qh,kh row-major f16 [N,64]; v transposed f16 [64,N].
// ---------------------------------------------------------------------------
__global__ __launch_bounds__(64) void gat_qkv_kernel(
    const float* __restrict__ x,
    const float* __restrict__ Wq, const float* __restrict__ bq,
    const float* __restrict__ Wk, const float* __restrict__ bk,
    const float* __restrict__ Wv, const float* __restrict__ bv,
    _Float16* __restrict__ qh, _Float16* __restrict__ kh, _Float16* __restrict__ vT)
{
    __shared__ float xs[D_IN];
    const int row = blockIdx.x;
    const int j   = threadIdx.x;                 // output feature 0..63
    xs[j]      = x[(size_t)row * D_IN + j];
    xs[j + 64] = x[(size_t)row * D_IN + 64 + j];
    __syncthreads();
    float aq = bq[j], ak = bk[j], av = bv[j];
    #pragma unroll 8
    for (int i = 0; i < D_IN; ++i) {
        float xv = xs[i];
        aq = fmaf(xv, Wq[i * D_HID + j], aq);
        ak = fmaf(xv, Wk[i * D_HID + j], ak);
        av = fmaf(xv, Wv[i * D_HID + j], av);
    }
    qh[(size_t)row * D_HID + j] = (_Float16)aq;
    kh[(size_t)row * D_HID + j] = (_Float16)ak;
    vT[(size_t)j * N_NODES + row] = (_Float16)av;
}

// ---------------------------------------------------------------------------
// Kernel B: bit-pack adjacency: 256MB int32 -> 8MB bitmask (wave32 ballot).
// ---------------------------------------------------------------------------
__global__ __launch_bounds__(256) void gat_pack_kernel(
    const int* __restrict__ a, uint32_t* __restrict__ packed)
{
    const int word = blockIdx.x * (blockDim.x >> 5) + (threadIdx.x >> 5);
    const int lane = threadIdx.x & 31;
    uint64_t b = __ballot(a[(size_t)word * 32 + lane] > 0);
    if (lane == 0) packed[word] = (uint32_t)b;
}

// ---------------------------------------------------------------------------
// Kernel C: fused flash-attention sweep, 64 keys per iteration.
// One wave owns a 16-row query tile; softmax bookkeeping amortized over
// four 16x16 score tiles per iteration.
// ---------------------------------------------------------------------------
__global__ __launch_bounds__(128) void gat_attn_kernel(
    const _Float16* __restrict__ qh, const _Float16* __restrict__ kh,
    const _Float16* __restrict__ vT, const uint32_t* __restrict__ adj,
    float* __restrict__ h)
{
    __shared__ _Float16 pbuf[4][16 * 64];        // per-wave P-transpose staging
    const int wave  = threadIdx.x >> 5;
    const int lane  = threadIdx.x & 31;
    const int n     = lane & 15;                 // column / A-row selector
    const int haf   = lane >> 4;                 // lane-half selector
    const int tile  = blockIdx.x * 4 + wave;     // 512 query tiles total
    const int qrow0 = tile * 16;
    _Float16* pt = pbuf[wave];

    // ---- Q as WMMA A-matrices (two K-chunks of D_HID), built once ----
    v16h aq0, aq1;
    {
        const _Float16* qp = qh + (size_t)(qrow0 + n) * D_HID;
        #pragma unroll
        for (int i = 0; i < 16; ++i) {
            int ka = i + haf * 16;
            aq0[i] = qp[ka];
            aq1[i] = qp[32 + ka];
        }
    }

    v8f o[4];                                    // O accumulator: 16x64 fp32
    #pragma unroll
    for (int nb = 0; nb < 4; ++nb) o[nb] = (v8f){};
    float mrow[8], lrow[8];
    #pragma unroll
    for (int r = 0; r < 8; ++r) { mrow[r] = NEG_BIG; lrow[r] = 0.0f; }

    const float scale_log2e = 1.4426950408889634f * 0.125f;  // log2(e)/sqrt(64)

    for (int kb = 0; kb < N_NODES / 64; ++kb) {
        const int key0 = kb * 64;

        // ---- scores: S = Q @ K^T for 64 keys (four 16x16 tiles) ----
        v8f c[4];
        #pragma unroll
        for (int t = 0; t < 4; ++t) {
            const _Float16* kp = kh + (size_t)(key0 + 16 * t + n) * D_HID + haf * 16;
            v16h blo = *(const v16h*)(kp);
            v16h bhi = *(const v16h*)(kp + 32);
            v8f cc = {};
            cc = WMMA_F16(aq0, blo, cc);
            cc = WMMA_F16(aq1, bhi, cc);
            c[t] = cc;
        }

        // ---- adjacency mask + online softmax (per C/D row slot) ----
        float corr[8];
        #pragma unroll
        for (int r = 0; r < 8; ++r) {
            const int row = qrow0 + r + haf * 8;
            const uint2 w = ((const uint2*)(adj + (size_t)row * WORDS_PER_ROW))[kb];
            float t0 = ((w.x >> n) & 1u)        ? c[0][r] * scale_log2e : NEG_BIG;
            float t1 = ((w.x >> (16 + n)) & 1u) ? c[1][r] * scale_log2e : NEG_BIG;
            float t2 = ((w.y >> n) & 1u)        ? c[2][r] * scale_log2e : NEG_BIG;
            float t3 = ((w.y >> (16 + n)) & 1u) ? c[3][r] * scale_log2e : NEG_BIG;
            float mx = fmaxf(fmaxf(t0, t1), fmaxf(t2, t3));
            #pragma unroll
            for (int d = 1; d < 16; d <<= 1) mx = fmaxf(mx, __shfl_xor(mx, d, 32));
            const float mnew = fmaxf(mrow[r], mx);
            const float cr = exp2f(mrow[r] - mnew);
            const float p0 = exp2f(t0 - mnew);
            const float p1 = exp2f(t1 - mnew);
            const float p2 = exp2f(t2 - mnew);
            const float p3 = exp2f(t3 - mnew);
            float rs = (p0 + p1) + (p2 + p3);
            #pragma unroll
            for (int d = 1; d < 16; d <<= 1) rs += __shfl_xor(rs, d, 32);
            mrow[r] = mnew;
            lrow[r] = lrow[r] * cr + rs;
            corr[r] = cr;
            _Float16* pw = pt + (r + haf * 8) * 64 + n;   // P row-major [16][64]
            pw[0]  = (_Float16)p0;
            pw[16] = (_Float16)p1;
            pw[32] = (_Float16)p2;
            pw[48] = (_Float16)p3;
        }

        // ---- rescale running O by exp correction ----
        #pragma unroll
        for (int r = 0; r < 8; ++r) {
            o[0][r] *= corr[r]; o[1][r] *= corr[r];
            o[2][r] *= corr[r]; o[3][r] *= corr[r];
        }

        // ---- gather P as two A-matrices (contiguous 16B LDS chunks) ----
        v16h pA0, pA1;
        {
            const _Float16* pr = pt + n * 64 + haf * 16;
            #pragma unroll
            for (int i = 0; i < 8; ++i) {
                pA0[i]     = pr[i];
                pA0[8 + i] = pr[8 + i];
                pA1[i]     = pr[32 + i];
                pA1[8 + i] = pr[40 + i];
            }
        }

        // ---- O += P @ V  (V pre-transposed -> contiguous B loads) ----
        #pragma unroll
        for (int nb = 0; nb < 4; ++nb) {
            const _Float16* vp = vT + (size_t)(nb * 16 + n) * N_NODES + key0 + haf * 16;
            v16h bv0 = *(const v16h*)(vp);
            v16h bv1 = *(const v16h*)(vp + 32);
            o[nb] = WMMA_F16(pA0, bv0, o[nb]);
            o[nb] = WMMA_F16(pA1, bv1, o[nb]);
        }
    }

    // ---- epilogue: normalize by l and store h (fp32) ----
    #pragma unroll
    for (int r = 0; r < 8; ++r) {
        const float inv = 1.0f / lrow[r];
        const size_t base = (size_t)(qrow0 + r + haf * 8) * D_HID + n;
        h[base]      = o[0][r] * inv;
        h[base + 16] = o[1][r] * inv;
        h[base + 32] = o[2][r] * inv;
        h[base + 48] = o[3][r] * inv;
    }
}

// ---------------------------------------------------------------------------
// Kernel D: out = mean_n( relu(h) @ Wo + bo )  -> 2 floats (single block).
// ---------------------------------------------------------------------------
__global__ __launch_bounds__(256) void gat_out_kernel(
    const float* __restrict__ h, const float* __restrict__ Wo,
    const float* __restrict__ bo, float* __restrict__ out)
{
    __shared__ float red0[256], red1[256];
    float a0 = 0.0f, a1 = 0.0f;
    for (int idx = threadIdx.x; idx < N_NODES * D_HID; idx += 256) {
        float v = h[idx];
        v = v > 0.0f ? v : 0.0f;
        const int j = idx & (D_HID - 1);
        a0 = fmaf(v, Wo[j * 2 + 0], a0);
        a1 = fmaf(v, Wo[j * 2 + 1], a1);
    }
    red0[threadIdx.x] = a0; red1[threadIdx.x] = a1;
    __syncthreads();
    for (int s = 128; s > 0; s >>= 1) {
        if (threadIdx.x < s) {
            red0[threadIdx.x] += red0[threadIdx.x + s];
            red1[threadIdx.x] += red1[threadIdx.x + s];
        }
        __syncthreads();
    }
    if (threadIdx.x == 0) {
        out[0] = red0[0] / (float)N_NODES + bo[0];
        out[1] = red1[0] / (float)N_NODES + bo[1];
    }
}

// ---------------------------------------------------------------------------
extern "C" void kernel_launch(void* const* d_in, const int* in_sizes, int n_in,
                              void* d_out, int out_size, void* d_ws, size_t ws_size,
                              hipStream_t stream) {
    const float* x  = (const float*)d_in[0];
    const int*   a  = (const int*)  d_in[1];
    const float* Wq = (const float*)d_in[2];
    const float* bq = (const float*)d_in[3];
    const float* Wk = (const float*)d_in[4];
    const float* bk = (const float*)d_in[5];
    const float* Wv = (const float*)d_in[6];
    const float* bv = (const float*)d_in[7];
    const float* Wo = (const float*)d_in[8];
    const float* bo = (const float*)d_in[9];

    char* ws = (char*)d_ws;
    _Float16* qh   = (_Float16*)(ws);                       // 1 MB
    _Float16* kh   = (_Float16*)(ws + (1u  << 20));         // 1 MB
    _Float16* vT   = (_Float16*)(ws + (2u  << 20));         // 1 MB
    uint32_t* adjp = (uint32_t*)(ws + (3u  << 20));         // 8 MB
    float*    h    = (float*)   (ws + (11u << 20));         // 2 MB
    float*    out  = (float*)d_out;

    gat_qkv_kernel<<<N_NODES, 64, 0, stream>>>(x, Wq, bq, Wk, bk, Wv, bv, qh, kh, vT);
    gat_pack_kernel<<<(N_NODES * WORDS_PER_ROW) / 8, 256, 0, stream>>>(a, adjp);
    gat_attn_kernel<<<(N_NODES / 16) / 4, 128, 0, stream>>>(qh, kh, vT, adjp, h);
    gat_out_kernel<<<1, 256, 0, stream>>>(h, Wo, bo, out);
}
